// Window_DMSAttention_10222022164490
// MI455X (gfx1250) — matrix operand order repaired and checked by reading, not compiled, not run on hardware
//
#include <hip/hip_runtime.h>

// ---- types ----
typedef _Float16 f16;
typedef __attribute__((ext_vector_type(16))) _Float16 v16h;
typedef __attribute__((ext_vector_type(8)))  float    v8f;

__device__ __forceinline__ v8f wmma16(v16h a, v16h b, v8f c) {
  // D = A(16x32 f16) x B(32x16 f16) + C(16x16 f32)
  return __builtin_amdgcn_wmma_f32_16x16x32_f16(false, a, false, b, (short)0, c, false, false);
}

// ---- problem constants ----
#define Bn   4
#define Cn   256
#define Hn   56
#define Wn   56
#define HWn  3136
#define HEADS 8
#define HCn  32
#define WSn  7
#define NWn  64
#define Pn   784
#define SCALEQ 0.0625f

// =====================================================================
// Kernel 1/6: 1x1 conv as GEMM  Out[b,o,p] = sum_c W[o,c]*X[b,c,p] + bias[o]
// one wave -> 4 o-tiles x one 16-wide p-tile; the expensive strided image
// B-fragment is loaded once per K-step and reused by 4 WMMAs.
// =====================================================================
__global__ __launch_bounds__(128) void k_conv1x1(const float* __restrict__ X,
                                                 const float* __restrict__ Wt,
                                                 const float* __restrict__ bias,
                                                 float* __restrict__ Out) {
  int wave = threadIdx.x >> 5;
  int lane = threadIdx.x & 31;
  int pt = blockIdx.x * 4 + wave;   // 0..195
  int og = blockIdx.y;              // 0..3 -> o-tiles og*4 .. og*4+3
  int b  = blockIdx.z;
  int p0 = pt * 16;
  const float* Xb = X + (size_t)b * Cn * HWn;
  bool hi = lane >= 16;
  int  l16 = lane & 15;
  v8f acc[4] = {v8f{}, v8f{}, v8f{}, v8f{}};
  for (int k0 = 0; k0 < 256; k0 += 32) {
    // B fragment (image): lane<16 col p0+l, K rows k0..k0+15; hi lanes k0+16..31
    const float* bp = Xb + (size_t)(k0 + (hi ? 16 : 0)) * HWn + (p0 + l16);
    v16h bf;
#pragma unroll
    for (int e = 0; e < 16; ++e) bf[e] = (f16)bp[(size_t)e * HWn];
#pragma unroll
    for (int i = 0; i < 4; ++i) {
      int o0 = (og * 4 + i) * 16;
      // A fragment (weights): lane<16 row o0+l, K k0..k0+15; hi lanes +16
      const float* ap = Wt + (o0 + l16) * 256 + k0 + (hi ? 16 : 0);
      v16h a;
#pragma unroll
      for (int e = 0; e < 8; ++e) { a[e] = (f16)ap[e]; a[8 + e] = (f16)ap[8 + e]; }
      acc[i] = wmma16(a, bf, acc[i]);
    }
  }
  int n = p0 + l16;
#pragma unroll
  for (int i = 0; i < 4; ++i) {
    int o0 = (og * 4 + i) * 16;
#pragma unroll
    for (int r = 0; r < 8; ++r) {
      int o = o0 + r + (hi ? 8 : 0);
      Out[((size_t)b * Cn + o) * HWn + n] = acc[i][r] + bias[o];
    }
  }
}

// =====================================================================
// Kernel 2/6: offset head. block = (window w, group n'=b*2+hg)
// dw7x7 stride7 + BN + exact GELU + pointwise(1568x128) + tanh -> pixel coords
// grid buffer: [n'][w][p][2]  ([0]=y_pix, [1]=x_pix)
// =====================================================================
__global__ __launch_bounds__(128) void k_offsets(const float* __restrict__ query,
                                                 const float* __restrict__ dw_w,
                                                 const float* __restrict__ bn_g,
                                                 const float* __restrict__ bn_b,
                                                 const float* __restrict__ pw_w,
                                                 float* __restrict__ grid) {
  __shared__ float act[128];
  int n = blockIdx.y;  // 0..7
  int w = blockIdx.x;  // 0..63
  int b = n >> 1, hg = n & 1;
  int wy = w >> 3, wx = w & 7;
  int t = threadIdx.x;
  if (t < 128) {
    int c = t;
    const float* qp = query + ((size_t)(b * Cn + hg * 128 + c)) * HWn + (wy * 7) * Wn + wx * 7;
    const float* ww = dw_w + c * 49;
    float s = 0.f;
#pragma unroll
    for (int i = 0; i < 7; ++i)
#pragma unroll
      for (int j = 0; j < 7; ++j) s += qp[i * Wn + j] * ww[i * 7 + j];
    s = s * (bn_g[c] * rsqrtf(1.f + 1e-5f)) + bn_b[c];
    s = 0.5f * s * (1.f + erff(s * 0.70710678118654752f));  // exact GELU
    act[c] = s;
  }
  __syncthreads();
  for (int j = t; j < 1568; j += 128) {
    const float* pw = pw_w + (size_t)j * 128;
    float s = 0.f;
    for (int c = 0; c < 128; ++c) s += pw[c] * act[c];
    float off = tanhf(s) * (2.0f / 56.0f);  // tanh * range * factor
    int d = (j >= 784) ? 1 : 0;
    int p = j - d * 784;
    int idx = d ? (p % 28) : (p / 28);                 // d=0 -> y uses ri, d=1 -> x uses rj
    float refv = ((float)(idx * 2) / 55.0f) * 2.0f - 1.0f;
    float gpix = (off + refv + 1.0f) * 0.5f * 55.0f;   // align_corners pixel space
    grid[(((size_t)n * NWn + w) * Pn + p) * 2 + d] = gpix;
  }
}

// =====================================================================
// Kernel 3/6: fused bilinear sampling + K/V projection per (b, window, ptile)
// TP=112 points/tile. gs tile (256 x 112, f16) lives in LDS only.
// Each wave caches all 8 A-fragments of one (proj, m-tile) in registers and
// sweeps the 7 N-tiles reading only LDS.
// kk layout: [b][w][head][p][hc]  (contiguous hc -> S-GEMM B-frag = 1 v16h load)
// vv layout: [b][w][head][hc][p]  (contiguous p  -> PV-GEMM B-frag = 1 v16h load)
// =====================================================================
#define TP  112
#define TPP 113
__global__ __launch_bounds__(256) void k_sample_kv(const float* __restrict__ X,
                                                   const float* __restrict__ grid,
                                                   const float* __restrict__ k_w,
                                                   const float* __restrict__ k_b,
                                                   const float* __restrict__ v_w,
                                                   const float* __restrict__ v_b,
                                                   f16* __restrict__ kk,
                                                   f16* __restrict__ vv) {
  __shared__ f16   gs[256 * TPP];
  __shared__ float w4[2][TP][4];
  __shared__ int   cxy[2][TP];
  int pt = blockIdx.x;  // 0..6
  int w  = blockIdx.y;  // 0..63
  int b  = blockIdx.z;
  int t = threadIdx.x;
  // ---- per-point bilinear weights (zeros padding, align_corners) ----
  if (t < 2 * TP) {
    int hg = t / TP, pl = t % TP;
    int p = pt * TP + pl;
    const float* g = grid + (((size_t)(b * 2 + hg) * NWn + w) * Pn + p) * 2;
    float y = g[0], x = g[1];
    float xf = floorf(x), yf = floorf(y);
    int x0 = (int)xf, y0 = (int)yf;
    float wx1 = x - xf, wx0 = 1.f - wx1;
    float wy1 = y - yf, wy0 = 1.f - wy1;
    float mx0 = (x0 >= 0 && x0 <= 55) ? 1.f : 0.f;
    float mx1 = (x0 + 1 >= 0 && x0 + 1 <= 55) ? 1.f : 0.f;
    float my0 = (y0 >= 0 && y0 <= 55) ? 1.f : 0.f;
    float my1 = (y0 + 1 >= 0 && y0 + 1 <= 55) ? 1.f : 0.f;
    w4[hg][pl][0] = wx0 * wy0 * mx0 * my0;
    w4[hg][pl][1] = wx1 * wy0 * mx1 * my0;
    w4[hg][pl][2] = wx0 * wy1 * mx0 * my1;
    w4[hg][pl][3] = wx1 * wy1 * mx1 * my1;
    int x0c = min(max(x0, 0), 55), x1c = min(max(x0 + 1, 0), 55);
    int y0c = min(max(y0, 0), 55), y1c = min(max(y0 + 1, 0), 55);
    cxy[hg][pl] = x0c | (x1c << 8) | (y0c << 16) | (y1c << 24);
  }
  __syncthreads();
  // ---- sample gs tile into LDS (f16) ----
  for (int s = t; s < 256 * TP; s += 256) {
    int pl = s % TP;
    int c  = s / TP;
    int hg = c >> 7;
    int pk = cxy[hg][pl];
    int x0c = pk & 0xFF, x1c = (pk >> 8) & 0xFF, y0c = (pk >> 16) & 0xFF, y1c = (pk >> 24) & 0xFF;
    const float* plane = X + ((size_t)(b * Cn + c)) * HWn;
    float v = w4[hg][pl][0] * plane[y0c * Wn + x0c] + w4[hg][pl][1] * plane[y0c * Wn + x1c] +
              w4[hg][pl][2] * plane[y1c * Wn + x0c] + w4[hg][pl][3] * plane[y1c * Wn + x1c];
    gs[c * TPP + pl] = (f16)v;
  }
  __syncthreads();
  // ---- K / V projection GEMM: 64 (proj,mtile) groups / 8 waves ----
  int wave = t >> 5, lane = t & 31;
  bool hi = lane >= 16;
  int  l16 = lane & 15;
  for (int g = wave; g < 64; g += 8) {
    int which = (g >= 32) ? 1 : 0;
    int mt = g & 31;
    const float* Wm = which ? v_w : k_w;
    const float* bs = which ? v_b : k_b;
    int o0 = mt * 16;
    // cache all 8 K-step A-fragments (weights) in registers
    v16h A[8];
#pragma unroll
    for (int kc = 0; kc < 8; ++kc) {
      const float* ap = Wm + (o0 + l16) * 256 + kc * 32 + (hi ? 16 : 0);
#pragma unroll
      for (int e = 0; e < 8; ++e) { A[kc][e] = (f16)ap[e]; A[kc][8 + e] = (f16)ap[8 + e]; }
    }
    for (int nt = 0; nt < 7; ++nt) {
      v8f acc = {};
#pragma unroll
      for (int kc = 0; kc < 8; ++kc) {
        const f16* gp = &gs[(kc * 32 + (hi ? 16 : 0)) * TPP + nt * 16 + l16];
        v16h bf;
#pragma unroll
        for (int e = 0; e < 16; ++e) bf[e] = gp[e * TPP];
        acc = wmma16(A[kc], bf, acc);
      }
      int p = pt * TP + nt * 16 + l16;
      size_t base = (size_t)(b * NWn + w) * HEADS;
#pragma unroll
      for (int r = 0; r < 8; ++r) {
        int o = o0 + r + (hi ? 8 : 0);
        float val = acc[r] + bs[o];
        int head = o >> 5, hc = o & 31;
        if (!which) kk[(base + head) * ((size_t)Pn * HCn) + (size_t)p * HCn + hc] = (f16)val;
        else        vv[(base + head) * ((size_t)Pn * HCn) + (size_t)hc * Pn + p] = (f16)val;
      }
    }
  }
}

// =====================================================================
// Kernel 4/6: flash attention. one wave per (head, window, b).
// Q(49pad64 x 32) vs 784 keys in 25 chunks of 32 (tail masked).
// =====================================================================
__global__ __launch_bounds__(32) void k_attn(const float* __restrict__ query,
                                             const f16* __restrict__ kk,
                                             const f16* __restrict__ vv,
                                             float* __restrict__ out) {
  __shared__ f16 Pl[64 * 34];
  int head = blockIdx.x, w = blockIdx.y, b = blockIdx.z;
  int wy = w >> 3, wx = w & 7;
  int lane = threadIdx.x;
  bool hi = lane >= 16;
  int  l16 = lane & 15;
  // ---- Q fragments (A layout), scaled by dim^-0.5, zero-padded rows 49..63 ----
  v16h Qf[4];
#pragma unroll
  for (int mt = 0; mt < 4; ++mt) {
    v16h a;
#pragma unroll
    for (int e = 0; e < 16; ++e) a[e] = (f16)0.f;
    int row = mt * 16 + l16;
    if (row < 49) {
      int hpix = wy * 7 + row / 7, xpix = wx * 7 + row % 7;
      const float* qp = query + ((size_t)(b * Cn + head * HCn)) * HWn + hpix * Wn + xpix;
      int kb = hi ? 16 : 0;
#pragma unroll
      for (int e = 0; e < 8; ++e) {
        a[e]     = (f16)(qp[(size_t)(kb + e) * HWn] * SCALEQ);
        a[8 + e] = (f16)(qp[(size_t)(kb + 8 + e) * HWn] * SCALEQ);
      }
    }
    Qf[mt] = a;
  }
  const f16* kkb = kk + ((size_t)(b * NWn + w) * HEADS + head) * ((size_t)Pn * HCn);
  const f16* vvb = vv + ((size_t)(b * NWn + w) * HEADS + head) * ((size_t)Pn * HCn);
  v8f O[4][2];
  float mi[4][8], li[4][8];
#pragma unroll
  for (int mt = 0; mt < 4; ++mt) {
    O[mt][0] = v8f{}; O[mt][1] = v8f{};
#pragma unroll
    for (int r = 0; r < 8; ++r) { mi[mt][r] = -1e30f; li[mt][r] = 0.f; }
  }
  v8f zc = {};
  for (int cidx = 0; cidx < 25; ++cidx) {
    int p0 = cidx * 32;
    // kk B-fragments: contiguous 16 halves per lane
    v16h kf0 = *(const v16h*)(kkb + (size_t)(p0 + l16) * HCn + (hi ? 16 : 0));
    v16h kf1 = *(const v16h*)(kkb + (size_t)(p0 + 16 + l16) * HCn + (hi ? 16 : 0));
    float mask0 = (p0 + l16 < Pn) ? 0.f : -1e30f;
    float mask1 = (p0 + 16 + l16 < Pn) ? 0.f : -1e30f;
#pragma unroll
    for (int mt = 0; mt < 4; ++mt) {
      v8f s0 = wmma16(Qf[mt], kf0, zc);
      v8f s1 = wmma16(Qf[mt], kf1, zc);
#pragma unroll
      for (int r = 0; r < 8; ++r) {
        float a0 = s0[r] + mask0, a1 = s1[r] + mask1;
        float mx = fmaxf(a0, a1);
        for (int d = 1; d < 16; d <<= 1) mx = fmaxf(mx, __shfl_xor(mx, d, 32));
        float mnew = fmaxf(mi[mt][r], mx);
        float sf = __expf(mi[mt][r] - mnew);
        float e0 = __expf(a0 - mnew), e1 = __expf(a1 - mnew);
        float rs = e0 + e1;
        for (int d = 1; d < 16; d <<= 1) rs += __shfl_xor(rs, d, 32);
        li[mt][r] = li[mt][r] * sf + rs;
        mi[mt][r] = mnew;
        O[mt][0][r] *= sf;
        O[mt][1][r] *= sf;
        int row = mt * 16 + r + (hi ? 8 : 0);
        Pl[row * 34 + l16]      = (f16)e0;
        Pl[row * 34 + 16 + l16] = (f16)e1;
      }
    }
    __syncthreads();
    // vv B-fragments: contiguous 16 halves per lane
    v16h vf0 = *(const v16h*)(vvb + (size_t)(0 * 16 + l16) * Pn + p0 + (hi ? 16 : 0));
    v16h vf1 = *(const v16h*)(vvb + (size_t)(1 * 16 + l16) * Pn + p0 + (hi ? 16 : 0));
#pragma unroll
    for (int mt = 0; mt < 4; ++mt) {
      const f16* pp = &Pl[(mt * 16 + l16) * 34 + (hi ? 16 : 0)];
      v16h pa;
#pragma unroll
      for (int e = 0; e < 8; ++e) { pa[e] = pp[e]; pa[8 + e] = pp[8 + e]; }
      O[mt][0] = wmma16(pa, vf0, O[mt][0]);
      O[mt][1] = wmma16(pa, vf1, O[mt][1]);
    }
    __syncthreads();
  }
  // ---- epilogue: normalize + window reverse ----
#pragma unroll
  for (int mt = 0; mt < 4; ++mt) {
#pragma unroll
    for (int r = 0; r < 8; ++r) {
      int row = mt * 16 + r + (hi ? 8 : 0);
      if (row < 49) {
        int hpix = wy * 7 + row / 7, xpix = wx * 7 + row % 7;
        float inv = 1.f / li[mt][r];
#pragma unroll
        for (int nh = 0; nh < 2; ++nh) {
          int hc = nh * 16 + l16;
          out[((size_t)(b * Cn + head * HCn + hc)) * HWn + hpix * Wn + xpix] = O[mt][nh][r] * inv;
        }
      }
    }
  }
}

// =====================================================================
// Kernel 5/6: depthwise 3x3 pos conv + bias + residual
// =====================================================================
__global__ __launch_bounds__(256) void k_pos(const float* __restrict__ attn,
                                             const float* __restrict__ pos_w,
                                             const float* __restrict__ pos_b,
                                             float* __restrict__ y) {
  int idx = blockIdx.x * 256 + threadIdx.x;
  if (idx >= Bn * Cn * HWn) return;
  int x = idx % Wn;
  int h = (idx / Wn) % Hn;
  int c = (idx / HWn) % Cn;
  int b = idx / (HWn * Cn);
  const float* pw = pos_w + c * 9;
  const float* ap = attn + ((size_t)(b * Cn + c)) * HWn;
  float s = pos_b[c];
#pragma unroll
  for (int di = -1; di <= 1; ++di)
#pragma unroll
    for (int dj = -1; dj <= 1; ++dj) {
      int hh = h + di, ww = x + dj;
      if (hh >= 0 && hh < Hn && ww >= 0 && ww < Wn) s += ap[hh * Wn + ww] * pw[(di + 1) * 3 + (dj + 1)];
    }
  y[idx] = ap[h * Wn + x] + s;
}

// =====================================================================
// launcher
// =====================================================================
extern "C" void kernel_launch(void* const* d_in, const int* in_sizes, int n_in,
                              void* d_out, int out_size, void* d_ws, size_t ws_size,
                              hipStream_t stream) {
  const float* x       = (const float*)d_in[0];
  const float* q_w     = (const float*)d_in[1];
  const float* q_b     = (const float*)d_in[2];
  const float* k_w     = (const float*)d_in[3];
  const float* k_b     = (const float*)d_in[4];
  const float* v_w     = (const float*)d_in[5];
  const float* v_b     = (const float*)d_in[6];
  const float* off_dw  = (const float*)d_in[7];
  const float* bn_g    = (const float*)d_in[8];
  const float* bn_b    = (const float*)d_in[9];
  const float* off_pw  = (const float*)d_in[10];
  const float* pos_w   = (const float*)d_in[11];
  const float* pos_b   = (const float*)d_in[12];
  const float* proj_w  = (const float*)d_in[13];
  const float* proj_b  = (const float*)d_in[14];
  float* out = (float*)d_out;

  char* ws = (char*)d_ws;
  size_t off = 0;
  auto alloc = [&](size_t bytes) { size_t o = off; off = (off + bytes + 255) & ~(size_t)255; return o; };
  float* query = (float*)(ws + alloc(sizeof(float) * (size_t)Bn * Cn * HWn));
  float* gridb = (float*)(ws + alloc(sizeof(float) * (size_t)8 * NWn * Pn * 2));
  float* attnb = (float*)(ws + alloc(sizeof(float) * (size_t)Bn * Cn * HWn));
  float* ybuf  = (float*)(ws + alloc(sizeof(float) * (size_t)Bn * Cn * HWn));
  f16* kk = (f16*)(ws + alloc(sizeof(f16) * ((size_t)Bn * NWn * HEADS * Pn * HCn + 1024)));
  f16* vv = (f16*)(ws + alloc(sizeof(f16) * ((size_t)Bn * NWn * HEADS * Pn * HCn + 1024)));

  k_conv1x1<<<dim3(49, 4, Bn), 128, 0, stream>>>(x, q_w, q_b, query);
  k_offsets<<<dim3(NWn, 8), 128, 0, stream>>>(query, off_dw, bn_g, bn_b, off_pw, gridb);
  k_sample_kv<<<dim3(7, NWn, Bn), 256, 0, stream>>>(x, gridb, k_w, k_b, v_w, v_b, kk, vv);
  k_attn<<<dim3(HEADS, NWn, Bn), 32, 0, stream>>>(query, kk, vv, attnb);
  k_pos<<<(Bn * Cn * HWn + 255) / 256, 256, 0, stream>>>(attnb, pos_w, pos_b, ybuf);
  k_conv1x1<<<dim3(49, 4, Bn), 128, 0, stream>>>(ybuf, proj_w, proj_b, out);
}
